// MultiHeadAttention_8435315769977
// MI455X (gfx1250) — matrix operand order repaired and checked by reading, not compiled, not run on hardware
//
#include <hip/hip_runtime.h>
#include <hip/hip_bf16.h>

#define B_  4
#define S_  2048
#define D_  1024
#define H_  16
#define DK_ 64
#define BSZ (B_ * S_)   // 8192 rows

typedef __attribute__((ext_vector_type(16))) __bf16 v16bf;
typedef __attribute__((ext_vector_type(8)))  float  v8f;
typedef __attribute__((ext_vector_type(4)))  unsigned v4u;
typedef __attribute__((ext_vector_type(8)))  unsigned v8u;

// ---------------------------------------------------------------- WMMA helper
__device__ __forceinline__ v8f wmma_bf16(v16bf a, v16bf b, v8f c) {
  // D = A(16x32 bf16) * B(32x16 bf16) + C(16x16 f32)
  return __builtin_amdgcn_wmma_f32_16x16x32_bf16(
      /*neg_a=*/false, a, /*neg_b=*/false, b,
      /*c_mod=*/(short)0, c, /*reuse_a=*/false, /*reuse_b=*/false);
}

union FragU { v16bf v; uint4 q[2]; };

// A-layout fragment: two contiguous 8-element (16B) chunks
__device__ __forceinline__ v16bf frag_2x8(const __bf16* p0, const __bf16* p1) {
  FragU u;
  u.q[0] = *(const uint4*)p0;
  u.q[1] = *(const uint4*)p1;
  return u.v;
}
// 16 contiguous elements (32B), k = 16h + s; this code's order for both A and B
__device__ __forceinline__ v16bf frag_16(const __bf16* p) {
  FragU u;
  u.q[0] = *(const uint4*)p;
  u.q[1] = *(const uint4*)(p + 8);
  return u.v;
}

// ------------------------------------------------ Tensor Data Mover (2D tile)
// Issue a TDM load of a 2D tile (tile_d1 rows x tile_d0 elements, row stride
// stride0 elements) from global to LDS.  data_size_log2: 1=2B, 2=4B.
// D# built per CDNA5 ISA ch.8; groups 2/3 zeroed (2D tensor, tile_dim2=0).
__device__ __forceinline__ void tdm_load_2d(unsigned lds_addr,
                                            unsigned long long gaddr,
                                            unsigned data_size_log2,
                                            unsigned tensor_d0, unsigned tensor_d1,
                                            unsigned tile_d0, unsigned tile_d1,
                                            unsigned stride0) {
  v4u g0;
  g0[0] = 1u;                                        // count=1, user descriptor
  g0[1] = lds_addr;                                  // LDS byte offset
  g0[2] = (unsigned)(gaddr & 0xffffffffull);         // global_addr[31:0]
  g0[3] = (unsigned)((gaddr >> 32) & 0x1ffffffull)   // global_addr[56:32]
        | (2u << 30);                                // type=2 ("image")
  v8u g1;
  g1[0] = (data_size_log2 << 16);                    // wg_mask=0, data_size
  g1[1] = (tensor_d0 & 0xffffu) << 16;               // tensor_dim0[15:0]
  g1[2] = (tensor_d0 >> 16) | ((tensor_d1 & 0xffffu) << 16);
  g1[3] = (tensor_d1 >> 16) | (tile_d0 << 16);       // tile_dim0
  g1[4] = tile_d1;                                   // tile_dim1, tile_dim2=0
  g1[5] = stride0;                                   // tensor_dim0_stride[31:0]
  g1[6] = 0u;
  g1[7] = 0u;
  v4u gz = {0u, 0u, 0u, 0u};
  asm volatile("tensor_load_to_lds %0, %1, %2, %3"
               :: "s"(g0), "s"(g1), "s"(gz), "s"(gz)
               : "memory");
}

__device__ __forceinline__ unsigned lds_offset_of(const void* p) {
  // Generic LDS address: low 32 bits are the LDS byte offset (ISA aperture map)
  return (unsigned)(size_t)p;
}

// ------------------------------------------------- fp32 -> bf16 weight repack
__global__ void __launch_bounds__(256)
cvt_f32_bf16_kernel(const float* __restrict__ x, __bf16* __restrict__ y, int n) {
  int i = blockIdx.x * 256 + threadIdx.x;
  if (i < n) y[i] = (__bf16)x[i];
}

// ------------------------------------------------------------- GEMM: X * W^T + b
// X: [M,K] fp32 (TDM-staged to LDS, converted to bf16 in-register),
// W: [N,K] bf16 (TDM-staged to LDS; B = W^T), Y: [M,N] fp32.
// Block: 128 threads (4 waves) -> 16x256 output tile; TDM double-buffers
// the 256x32 B tile (16KB) and 16x32 A tile (2KB) per K-step.
__global__ void __launch_bounds__(128)
gemm_xwT_bias_kernel(const float* __restrict__ X, const __bf16* __restrict__ Wb,
                     const float* __restrict__ bias, float* __restrict__ Y,
                     int M, int N, int K) {
  __shared__ __bf16 ldsB[2][256 * 32];   // [buf][row*32 + kx]
  __shared__ float  ldsA[2][16 * 32];    // [buf][row*32 + kx]

  const int lane = threadIdx.x & 31;
  const int wave = threadIdx.x >> 5;
  const int nl = lane & 15, hi = lane >> 4;
  const int m0 = blockIdx.x * 16;
  const int n0blk = blockIdx.y * 256;

  // prime buffer 0
  if (wave == 0) {
    tdm_load_2d(lds_offset_of(&ldsA[0][0]),
                (unsigned long long)(size_t)(X + (size_t)m0 * K),
                /*ds=*/2u, /*tensor*/ (unsigned)K, (unsigned)M,
                /*tile*/ 32u, 16u, (unsigned)K);
    tdm_load_2d(lds_offset_of(&ldsB[0][0]),
                (unsigned long long)(size_t)(Wb + (size_t)n0blk * K),
                /*ds=*/1u, /*tensor*/ (unsigned)K, (unsigned)N,
                /*tile*/ 32u, 256u, (unsigned)K);
  }

  v8f c[4] = {};
  int cur = 0;
  for (int kb = 0; kb < K; kb += 32) {
    if (wave == 0) __builtin_amdgcn_s_wait_tensorcnt(0);
    __syncthreads();   // tile in LDS visible to all; prev tile fully consumed
    if (wave == 0 && kb + 32 < K) {
      const int nxt = cur ^ 1;
      tdm_load_2d(lds_offset_of(&ldsA[nxt][0]),
                  (unsigned long long)(size_t)(X + (size_t)m0 * K + kb + 32),
                  2u, (unsigned)K, (unsigned)M, 32u, 16u, (unsigned)K);
      tdm_load_2d(lds_offset_of(&ldsB[nxt][0]),
                  (unsigned long long)(size_t)(Wb + (size_t)n0blk * K + kb + 32),
                  1u, (unsigned)K, (unsigned)N, 32u, 256u, (unsigned)K);
    }

    // load all four B fragments first (independent regs -> ds_load clause),
    // then the A tile, then drain with 4 back-to-back WMMAs.
    const __bf16* brow = &ldsB[cur][(wave * 64 + nl) * 32 + hi * 16];
    const v16bf b0 = frag_16(brow);
    const v16bf b1 = frag_16(brow + 16 * 32);
    const v16bf b2 = frag_16(brow + 32 * 32);
    const v16bf b3 = frag_16(brow + 48 * 32);

    // A fragment from LDS fp32 -> bf16
    v16bf a;
    {
      const float* pa = &ldsA[cur][nl * 32 + hi * 16];
#pragma unroll
      for (int e = 0; e < 8; ++e) a[e] = (__bf16)pa[e];
#pragma unroll
      for (int e = 0; e < 8; ++e) a[8 + e] = (__bf16)pa[8 + e];
    }

    c[0] = wmma_bf16(a, b0, c[0]);
    c[1] = wmma_bf16(a, b1, c[1]);
    c[2] = wmma_bf16(a, b2, c[2]);
    c[3] = wmma_bf16(a, b3, c[3]);
    cur ^= 1;
  }

  const int n0 = n0blk + wave * 64;
#pragma unroll
  for (int t = 0; t < 4; ++t) {
    const int n = n0 + t * 16 + nl;
    const float bv = bias[n];
#pragma unroll
    for (int r = 0; r < 8; ++r)
      Y[(size_t)(m0 + r + 8 * hi) * N + n] = c[t][r] + bv;
  }
}

// ------------------------------------------- RoPE + repack to [B,H,S,dk] bf16
__global__ void __launch_bounds__(256)
rope_pack_kernel(const float* __restrict__ P, __bf16* __restrict__ R, float outScale) {
  size_t i = (size_t)blockIdx.x * 256 + threadIdx.x;   // over B*S*D
  const int d = (int)(i & 63);
  const int h = (int)((i >> 6) & 15);
  const int s = (int)((i >> 10) & 2047);
  const int b = (int)(i >> 21);
  const float x   = P[i];
  const float rot = (d < 32) ? -P[i + 32] : P[i - 32];
  // theta = s * ROPE_BASE^(-(d%32)/32) ; log2(10000)=13.2877123795
  const float th = (float)s * exp2f(-(float)(d & 31) * (13.287712379549449f / 32.0f));
  float sn, cs;
  __sincosf(th, &sn, &cs);
  const float y = (x * cs + rot * sn) * outScale;
  R[((size_t)(b * H_ + h) * S_ + s) * DK_ + d] = (__bf16)y;
}

// --------------------------------- V repack transposed to [B,H,dk,S] bf16
__global__ void __launch_bounds__(256)
vtrans_pack_kernel(const float* __restrict__ P, __bf16* __restrict__ Vt) {
  size_t i = (size_t)blockIdx.x * 256 + threadIdx.x;
  const int d = (int)(i & 63);
  const int h = (int)((i >> 6) & 15);
  const int s = (int)((i >> 10) & 2047);
  const int b = (int)(i >> 21);
  Vt[((size_t)(b * H_ + h) * DK_ + d) * S_ + s] = (__bf16)P[i];
}

// ----------------------------------------------------- causal flash attention
// Wave owns 16 query rows of one (b,h); streams 32-key blocks to the diagonal.
__global__ void __launch_bounds__(128)
flash_attn_kernel(const __bf16* __restrict__ Qr, const __bf16* __restrict__ Kr,
                  const __bf16* __restrict__ Vt, float* __restrict__ ctx) {
  __shared__ __bf16 lds[4 * 16 * 32];   // per-wave 16x32 P staging (1KB each)
  const int lane = threadIdx.x & 31;
  const int wave = threadIdx.x >> 5;
  const int nl = lane & 15, hi = lane >> 4;

  const int tile = blockIdx.x * 4 + wave;
  const int bh = tile >> 7;              // S/16 = 128 query tiles per (b,h)
  const int q0 = (tile & 127) * 16;
  const int b = bh >> 4, h = bh & 15;

  const __bf16* Qb = Qr + (size_t)bh * S_ * DK_;
  const __bf16* Kb = Kr + (size_t)bh * S_ * DK_;
  const __bf16* Vb = Vt + (size_t)bh * DK_ * S_;
  __bf16* pbuf = lds + wave * (16 * 32);

  // Q fragments (A-layout, scale 1/8 already folded in at RoPE time)
  const __bf16* qrow = Qb + (size_t)(q0 + nl) * DK_;
  const v16bf qa0 = frag_16(qrow + hi * 16);
  const v16bf qa1 = frag_16(qrow + 32 + hi * 16);

  v8f o0 = {}, o1 = {}, o2 = {}, o3 = {};
  float m8[8], l8[8];
#pragma unroll
  for (int r = 0; r < 8; ++r) { m8[r] = -1e30f; l8[r] = 0.0f; }

  const int kb_max = (q0 + 15) >> 5;
  for (int kb = 0; kb <= kb_max; ++kb) {
    const int k0 = kb << 5;

    // scores S = Q K^T : load all four K fragments (clustered), then 4 WMMAs
    const __bf16* krow0 = Kb + (size_t)(k0 + nl) * DK_ + hi * 16;
    const __bf16* krow1 = Kb + (size_t)(k0 + 16 + nl) * DK_ + hi * 16;
    const v16bf kf00 = frag_16(krow0);
    const v16bf kf01 = frag_16(krow0 + 32);
    const v16bf kf10 = frag_16(krow1);
    const v16bf kf11 = frag_16(krow1 + 32);
    v8f s0v = {}, s1v = {};
    s0v = wmma_bf16(qa0, kf00, s0v);
    s0v = wmma_bf16(qa1, kf01, s0v);
    s1v = wmma_bf16(qa0, kf10, s1v);
    s1v = wmma_bf16(qa1, kf11, s1v);

    if (kb < kb_max) {  // prefetch next K/V block (global_prefetch_b8)
      __builtin_prefetch(Kb + (size_t)(k0 + 32 + nl) * DK_, 0, 1);
      __builtin_prefetch(Vb + (size_t)nl * S_ + k0 + 32, 0, 1);
    }

    const bool diag = (kb == kb_max);
    const int key0 = k0 + nl, key1 = k0 + 16 + nl;
    float alph[8];
#pragma unroll
    for (int r = 0; r < 8; ++r) {
      const int row = q0 + r + 8 * hi;
      float s0 = s0v[r], s1 = s1v[r];
      if (diag) {
        if (key0 > row) s0 = -1e30f;
        if (key1 > row) s1 = -1e30f;
      }
      // row max across the 16 N-lanes of each half-wave
      float rm = fmaxf(s0, s1);
      rm = fmaxf(rm, __shfl_xor(rm, 1, 16));
      rm = fmaxf(rm, __shfl_xor(rm, 2, 16));
      rm = fmaxf(rm, __shfl_xor(rm, 4, 16));
      rm = fmaxf(rm, __shfl_xor(rm, 8, 16));
      const float mold = m8[r];
      const float mnew = fmaxf(mold, rm);
      const float a = __expf(mold - mnew);
      const float p0 = __expf(s0 - mnew);
      const float p1 = __expf(s1 - mnew);
      float rs = p0 + p1;
      rs += __shfl_xor(rs, 1, 16);
      rs += __shfl_xor(rs, 2, 16);
      rs += __shfl_xor(rs, 4, 16);
      rs += __shfl_xor(rs, 8, 16);
      l8[r] = l8[r] * a + rs;
      m8[r] = mnew;
      alph[r] = a;
      // stage P (D-layout element -> row-major 16x32 in LDS)
      const int rl = r + 8 * hi;
      pbuf[rl * 32 + nl]      = (__bf16)p0;
      pbuf[rl * 32 + 16 + nl] = (__bf16)p1;
    }
    // rescale running output accumulators
#pragma unroll
    for (int r = 0; r < 8; ++r) {
      o0[r] *= alph[r]; o1[r] *= alph[r]; o2[r] *= alph[r]; o3[r] *= alph[r];
    }

    // load all four V fragments up front (clustered global_load_b128s)
    const __bf16* vrow = Vb + (size_t)nl * S_ + k0 + hi * 16;
    const v16bf vf0 = frag_16(vrow);
    const v16bf vf1 = frag_16(vrow + (size_t)16 * S_);
    const v16bf vf2 = frag_16(vrow + (size_t)32 * S_);
    const v16bf vf3 = frag_16(vrow + (size_t)48 * S_);

    asm volatile("s_wait_dscnt 0" ::: "memory");
    // read P back in A-layout (16x32, K = key index)
    const __bf16* prow = pbuf + nl * 32;
    const v16bf pa = frag_16(prow + hi * 16);

    // O += P * V : V^T stored [dk, S], B-lane = output channel, contiguous in S
    o0 = wmma_bf16(pa, vf0, o0);
    o1 = wmma_bf16(pa, vf1, o1);
    o2 = wmma_bf16(pa, vf2, o2);
    o3 = wmma_bf16(pa, vf3, o3);
  }

  // normalize and write context fp32 into [B,S,D] (col = h*64 + d)
#pragma unroll
  for (int r = 0; r < 8; ++r) {
    const float inv = 1.0f / l8[r];
    const size_t base = (size_t)(b * S_ + q0 + r + 8 * hi) * D_ + h * DK_;
    ctx[base + 0 * 16 + nl] = o0[r] * inv;
    ctx[base + 1 * 16 + nl] = o1[r] * inv;
    ctx[base + 2 * 16 + nl] = o2[r] * inv;
    ctx[base + 3 * 16 + nl] = o3[r] * inv;
  }
}

// ---------------------------------------------------------------------- launch
extern "C" void kernel_launch(void* const* d_in, const int* in_sizes, int n_in,
                              void* d_out, int out_size, void* d_ws, size_t ws_size,
                              hipStream_t stream) {
  (void)in_sizes; (void)n_in; (void)out_size; (void)ws_size;
  const float* q  = (const float*)d_in[0];
  const float* k  = (const float*)d_in[1];
  const float* v  = (const float*)d_in[2];
  const float* Wq = (const float*)d_in[3];
  const float* bq = (const float*)d_in[4];
  const float* Wk = (const float*)d_in[5];
  const float* bk = (const float*)d_in[6];
  const float* Wv = (const float*)d_in[7];
  const float* bv = (const float*)d_in[8];
  const float* Wo = (const float*)d_in[9];
  const float* bo = (const float*)d_in[10];
  float* out = (float*)d_out;

  char* w = (char*)d_ws;
  const size_t MB = 1ull << 20;
  __bf16* wqB = (__bf16*)(w + 0 * MB);   // 2 MB each
  __bf16* wkB = (__bf16*)(w + 2 * MB);
  __bf16* wvB = (__bf16*)(w + 4 * MB);
  __bf16* woB = (__bf16*)(w + 6 * MB);
  float*  proj = (float*)(w + 8 * MB);   // 32 MB fp32 projection scratch (reused)
  __bf16* Qr  = (__bf16*)(w + 40 * MB);  // 16 MB [B,H,S,dk]
  __bf16* Kr  = (__bf16*)(w + 56 * MB);  // 16 MB [B,H,S,dk]
  __bf16* Vt  = (__bf16*)(w + 72 * MB);  // 16 MB [B,H,dk,S]
  float*  ctx = (float*)(w + 88 * MB);   // 32 MB fp32 context

  const int DD = D_ * D_;
  dim3 gW((DD + 255) / 256);
  cvt_f32_bf16_kernel<<<gW, 256, 0, stream>>>(Wq, wqB, DD);
  cvt_f32_bf16_kernel<<<gW, 256, 0, stream>>>(Wk, wkB, DD);
  cvt_f32_bf16_kernel<<<gW, 256, 0, stream>>>(Wv, wvB, DD);
  cvt_f32_bf16_kernel<<<gW, 256, 0, stream>>>(Wo, woB, DD);

  dim3 gG(BSZ / 16, D_ / 256);                         // (512, 4)
  dim3 gE((unsigned)((size_t)BSZ * D_ / 256));         // 32768

  // Q path (scale 1/sqrt(dk) folded into Q)
  gemm_xwT_bias_kernel<<<gG, 128, 0, stream>>>(q, wqB, bq, proj, BSZ, D_, D_);
  rope_pack_kernel<<<gE, 256, 0, stream>>>(proj, Qr, 0.125f);
  // K path
  gemm_xwT_bias_kernel<<<gG, 128, 0, stream>>>(k, wkB, bk, proj, BSZ, D_, D_);
  rope_pack_kernel<<<gE, 256, 0, stream>>>(proj, Kr, 1.0f);
  // V path
  gemm_xwT_bias_kernel<<<gG, 128, 0, stream>>>(v, wvB, bv, proj, BSZ, D_, D_);
  vtrans_pack_kernel<<<gE, 256, 0, stream>>>(proj, Vt);

  // attention: B*H*(S/16) = 8192 wave-tiles, 4 waves/block
  flash_attn_kernel<<<dim3(B_ * H_ * (S_ / 16) / 4), 128, 0, stream>>>(Qr, Kr, Vt, ctx);

  // output projection -> d_out
  gemm_xwT_bias_kernel<<<gG, 128, 0, stream>>>(ctx, woB, bo, out, BSZ, D_, D_);
}
